// PsychoAcousticLoss_34797825032622
// MI455X (gfx1250) — matrix operand order repaired and checked by reading, not compiled, not run on hardware
//
#include <hip/hip_runtime.h>

// ---------------- constants ----------------
#define NFFT     2048
#define HOP      512
#define NFRAMES  4097          // 1 + (L + 2*pad - NFFT)/HOP
#define NBINS    1025          // NFFT/2 + 1
#define NFREQS   1024          // NFFT/2
#define NPAD     1088          // NBINS padded to 17*64
#define LSIG     2097152
#define PAD      1024
#define NFILTS   64

typedef __attribute__((ext_vector_type(16))) _Float16 v16h;
typedef __attribute__((ext_vector_type(8)))  float    v8f;

union FragU { uint4 u[2]; v16h v; };

__device__ inline v8f wmma_f16(v16h a, v16h b, v8f c) {
  // (neg_a, A, neg_b, B, c_mod, C, reuse_a, reuse_b)
  return __builtin_amdgcn_wmma_f32_16x16x32_f16(false, a, false, b, (short)0, c, false, false);
}

// Wave-relative LDS byte address (generic shared ptr low 32 bits == AS3 offset on amdgcn).
__device__ inline uint32_t lds_addr_of(const void* p) {
  return (uint32_t)(uintptr_t)p;
}

// Async 16-byte global -> LDS copy, tracked by ASYNCcnt.
__device__ inline void async_copy_b128(uint32_t lds_byte_addr, const void* gsrc) {
  const unsigned long long g = (unsigned long long)(uintptr_t)gsrc;
  asm volatile("global_load_async_to_lds_b128 %0, %1, off"
               :: "v"(lds_byte_addr), "v"(g) : "memory");
}

__device__ inline void wait_async_all() {
  asm volatile("s_wait_asynccnt 0x0" ::: "memory");
}

// A fragment (16x32 f16) from row-major LDS tile with stride 32.
// Lane l: m = l&15, half h = l>>4; element e: k = h*16+e (contiguous).
__device__ inline v16h frag_a_from_lds(const _Float16* As, int mo, int lane) {
  const int m = lane & 15, h = lane >> 4;
  const _Float16* row = As + (mo + m) * 32;
  FragU r;
  r.u[0] = *reinterpret_cast<const uint4*>(row + h * 16);
  r.u[1] = *reinterpret_cast<const uint4*>(row + h * 16 + 8);
  return r.v;
}

// B fragment (32x16 f16) from column-major LDS tile (Bs[n][k], 32 k-values contiguous per n).
// Lane l: n = l&15, kbase = (l>>4)*16; element e: k = kbase + e (contiguous).
__device__ inline v16h frag_b_from_lds(const _Float16* Bs, int no, int lane) {
  const int n = lane & 15, kb = (lane >> 4) << 4;
  const _Float16* p = Bs + (no + n) * 32 + kb;
  FragU r;
  r.u[0] = *reinterpret_cast<const uint4*>(p);
  r.u[1] = *reinterpret_cast<const uint4*>(p + 8);
  return r.v;
}

// ---------------- constant builders ----------------
// Cosine matrices stored pre-swizzled in the exact per-K-step LDS tile layout:
//   CfS[kt][n][kk] = cos-fwd(k = kt*32+kk, n)   (kt over 2048/32, n over NPAD)
//   CiS[kt][n][kk] = cos-inv(k = kt*32+kk, n)   (kt over NPAD/32, n over 2048)
// so a (ks, n0) B-tile is a contiguous 64*32-half (4KB) block -> one async DMA per K-step.
__global__ void k_build_mats(_Float16* __restrict__ CfS, _Float16* __restrict__ CiS) {
  const int idx = blockIdx.x * 256 + threadIdx.x;
  const float w0 = 0.003067961575771282f; // 2*pi/2048
  if (idx < NFFT * NPAD) {
    const int kk = idx & 31;
    const int rest = idx >> 5;
    const int n  = rest % NPAD;
    const int kt = rest / NPAD;
    const int k  = kt * 32 + kk;                    // time index
    float v = 0.f;
    if (n < NBINS) { const int m = (k * n) & (NFFT - 1); v = __builtin_cosf((float)m * w0); }
    CfS[idx] = (_Float16)v;
  }
  if (idx < NPAD * NFFT) {
    const int kk = idx & 31;
    const int rest = idx >> 5;
    const int n  = rest & (NFFT - 1);
    const int kt = rest >> 11;
    const int k  = kt * 32 + kk;                    // frequency index
    float v = 0.f;
    if (k < NBINS) {
      const float wk = (k == 0 || k == NFREQS) ? 1.f : 2.f;
      const int m = (k * n) & (NFFT - 1);
      v = wk * __builtin_cosf((float)m * w0);       // 1/NFFT folded into OLA kernel
    }
    CiS[idx] = (_Float16)v;
  }
}

__global__ void k_build_bark(int* __restrict__ filt, int* __restrict__ fstart,
                             float* __restrict__ scale, float* __restrict__ bq,
                             float* __restrict__ sprA) {
  const int tid = threadIdx.x;
  __shared__ int   sfilt[NBINS];
  __shared__ float sfv[128];
  const float maxbark = 6.f * asinhf(40.f);       // hz2bark(24000)
  const float step    = maxbark / (NFILTS - 1);
  for (int n = tid; n < NBINS; n += 128) {
    const float freq = (float)n * (48000.f / (float)NFFT);
    const float bb   = 6.f * asinhf(freq / 600.f);
    int j = (int)rintf(bb / step);
    j = min(max(j, 0), NFILTS - 1);
    sfilt[n] = j; filt[n] = j;
  }
  __syncthreads();
  if (tid < NFILTS) {                              // W row-sum -> W_inv scale
    int c = 0;
    for (int n = 0; n < NBINS; ++n) c += (sfilt[n] == tid);
    scale[tid] = sqrtf(1.f / ((float)c + 1e-6f));
  }
  if (tid < NFILTS + 1) {                          // filt is monotone: contiguous ranges
    int s = NBINS;
    for (int n = 0; n < NBINS; ++n) if (sfilt[n] >= tid) { s = n; break; }
    fstart[tid] = s;
  }
  const float ALPHA = 0.8f;
  if (tid < 128) {                                 // sf_volt
    float db;
    if (tid < 64) db = -maxbark * 27.f + (float)tid * ((-8.f + maxbark * 27.f) / 63.f) - 23.5f;
    else          db = (float)(tid - 64) * (-maxbark * 12.f / 63.f) - 23.5f;
    sfv[tid] = powf(10.f, db * (0.05f * ALPHA));
  }
  __syncthreads();
  for (int idx = tid; idx < NFILTS * NFILTS; idx += 128) {
    const int j = idx >> 6, j2 = idx & 63;         // sprmat_a[j][j2] = sf_volt[64-j+j2]^alpha
    sprA[idx] = powf(sfv[NFILTS - j + j2], ALPHA);
  }
  if (tid < NFILTS) {                              // threshold-in-quiet
    const float f  = 600.f * sinhf((float)tid * step / 6.f) + 1e-6f;
    const float fk = f * 0.001f;
    float LTQ = 3.64f * powf(fk, -0.8f)
              - 6.5f * expf(-0.6f * (fk - 3.3f) * (fk - 3.3f))
              + 0.001f * powf(fk, 4.f);
    LTQ = fminf(fmaxf(LTQ, -20.f), 120.f);
    bq[tid] = powf(10.f, (LTQ - 60.f) * 0.05f);
  }
}

// ---------------- forward GEMM: spec = frames(x) @ Cf ----------------
__global__ __launch_bounds__(256) void k_gemm_fwd(const float* __restrict__ x,
                                                  const _Float16* __restrict__ CfS,
                                                  float* __restrict__ spec) {
  __shared__ alignas(16) _Float16 As[128 * 32];   // row-major [m][k]
  __shared__ alignas(16) _Float16 Bs[64 * 32];    // col-major [n][k]
  const int tid  = threadIdx.x;
  const int lane = tid & 31, w = tid >> 5;
  const int wm = w >> 1, wn = w & 1;              // 4x2 wave grid, 32x32 per wave
  const int f0 = blockIdx.y * 128, n0 = blockIdx.x * 64;
  const uint32_t BsAddr = lds_addr_of(&Bs[0]);
  v8f acc[2][2] = {};

  for (int ks = 0; ks < NFFT; ks += 32) {
    __syncthreads();
    // stage B: contiguous pre-swizzled 4KB tile -> LDS via async DMA (ASYNCcnt)
    async_copy_b128(BsAddr + tid * 16,
                    CfS + ((size_t)(ks >> 5) * NPAD + n0) * 32 + tid * 8);
    { // stage A: frame rows generated on the fly with reflect padding
      const int m = tid >> 1, c0 = (tid & 1) << 4;
      const int f = f0 + m;
      _Float16* dst = As + m * 32 + c0;
      if (f < NFRAMES) {
        const int base = f * HOP + ks + c0 - PAD;
        #pragma unroll
        for (int c = 0; c < 16; ++c) {
          int j = base + c;
          j = (j < 0) ? -j : j;
          j = (j >= LSIG) ? (2 * LSIG - 2 - j) : j;
          dst[c] = (_Float16)x[j];
        }
      } else {
        #pragma unroll
        for (int c = 0; c < 16; ++c) dst[c] = (_Float16)0.f;
      }
    }
    wait_async_all();
    __syncthreads();
    const v16h a0 = frag_a_from_lds(As, wm * 32,      lane);
    const v16h a1 = frag_a_from_lds(As, wm * 32 + 16, lane);
    const v16h b0 = frag_b_from_lds(Bs, wn * 32,      lane);
    const v16h b1 = frag_b_from_lds(Bs, wn * 32 + 16, lane);
    acc[0][0] = wmma_f16(a0, b0, acc[0][0]);
    acc[0][1] = wmma_f16(a0, b1, acc[0][1]);
    acc[1][0] = wmma_f16(a1, b0, acc[1][0]);
    acc[1][1] = wmma_f16(a1, b1, acc[1][1]);
  }

  const int mrow = (lane >> 4) << 3;
  const int ncol = lane & 15;
  #pragma unroll
  for (int i = 0; i < 2; ++i)
    #pragma unroll
    for (int jj = 0; jj < 2; ++jj) {
      const int nn = n0 + wn * 32 + jj * 16 + ncol;
      #pragma unroll
      for (int r = 0; r < 8; ++r) {
        const int f = f0 + wm * 32 + i * 16 + mrow + r;
        if (f < NFRAMES && nn < NBINS)
          spec[(size_t)f * NPAD + nn] = acc[i][jj][r];
      }
    }
}

// ---------------- bark masking: masked = spec / mT ----------------
__global__ __launch_bounds__(128) void k_mask(const float* __restrict__ spec,
                                              _Float16* __restrict__ masked,
                                              const int* __restrict__ filt,
                                              const int* __restrict__ fstart,
                                              const float* __restrict__ scale,
                                              const float* __restrict__ bq,
                                              const float* __restrict__ sprA) {
  const int f = blockIdx.x, tid = threadIdx.x;
  __shared__ float srow[NPAD];
  __shared__ float mXa[NFILTS];
  __shared__ float mtb[NFILTS];
  const float* sp = spec + (size_t)f * NPAD;
  for (int n = tid; n < NPAD; n += 128) srow[n] = (n < NBINS) ? sp[n] : 0.f;
  __syncthreads();
  if (tid < NFILTS) {          // deterministic per-filter energy (contiguous bin ranges)
    const int s = fstart[tid];
    const int e = min(fstart[tid + 1], NFREQS);
    float acc = 0.f;
    for (int n = s; n < e; ++n) { const float v = srow[n]; acc += v * v; }
    mXa[tid] = powf(sqrtf(acc), 0.8f);           // mXbark^alpha
  }
  __syncthreads();
  if (tid < NFILTS) {
    float t = 0.f;
    for (int j = 0; j < NFILTS; ++j) t += mXa[j] * sprA[(j << 6) + tid];
    float m = powf(t, 1.25f);                    // ^(1/alpha)
    m = fmaxf(m, bq[tid]);
    mtb[tid] = m * scale[tid];                   // fold W_inv scale
  }
  __syncthreads();
  _Float16* mk = masked + (size_t)f * NPAD;
  for (int n = tid; n < NPAD; n += 128) {
    float o = 0.f;
    if (n < NBINS) o = srow[n] / mtb[filt[n]];
    mk[n] = (_Float16)o;
  }
}

// ---------------- inverse GEMM: invf = masked @ Ci ----------------
__global__ __launch_bounds__(256) void k_gemm_inv(const _Float16* __restrict__ masked,
                                                  const _Float16* __restrict__ CiS,
                                                  float* __restrict__ invf) {
  __shared__ alignas(16) _Float16 As[128 * 32];
  __shared__ alignas(16) _Float16 Bs[64 * 32];
  const int tid  = threadIdx.x;
  const int lane = tid & 31, w = tid >> 5;
  const int wm = w >> 1, wn = w & 1;
  const int f0 = blockIdx.y * 128, n0 = blockIdx.x * 64;
  const uint32_t AsAddr = lds_addr_of(&As[0]);
  const uint32_t BsAddr = lds_addr_of(&Bs[0]);
  v8f acc[2][2] = {};

  for (int ks = 0; ks < NPAD; ks += 32) {
    __syncthreads();
    // stage B: contiguous pre-swizzled 4KB tile -> LDS via async DMA
    async_copy_b128(BsAddr + tid * 16,
                    CiS + ((size_t)(ks >> 5) * NFFT + n0) * 32 + tid * 8);
    { // stage A: half-row (32B) per lane from the f16 masked spectrum via async DMA
      const int m = tid >> 1, h = tid & 1;
      const int f = f0 + m;
      const uint32_t dstA = AsAddr + (uint32_t)(m * 64 + h * 32);
      if (f < NFRAMES) {
        const _Float16* src = masked + (size_t)f * NPAD + ks + h * 16;
        async_copy_b128(dstA,       src);
        async_copy_b128(dstA + 16u, src + 8);
      } else {
        uint4* z = reinterpret_cast<uint4*>(As + m * 32 + h * 16);
        z[0] = make_uint4(0, 0, 0, 0);
        z[1] = make_uint4(0, 0, 0, 0);
      }
    }
    wait_async_all();
    __syncthreads();
    const v16h a0 = frag_a_from_lds(As, wm * 32,      lane);
    const v16h a1 = frag_a_from_lds(As, wm * 32 + 16, lane);
    const v16h b0 = frag_b_from_lds(Bs, wn * 32,      lane);
    const v16h b1 = frag_b_from_lds(Bs, wn * 32 + 16, lane);
    acc[0][0] = wmma_f16(a0, b0, acc[0][0]);
    acc[0][1] = wmma_f16(a0, b1, acc[0][1]);
    acc[1][0] = wmma_f16(a1, b0, acc[1][0]);
    acc[1][1] = wmma_f16(a1, b1, acc[1][1]);
  }

  const int mrow = (lane >> 4) << 3;
  const int ncol = lane & 15;
  #pragma unroll
  for (int i = 0; i < 2; ++i)
    #pragma unroll
    for (int jj = 0; jj < 2; ++jj) {
      const int nn = n0 + wn * 32 + jj * 16 + ncol;
      #pragma unroll
      for (int r = 0; r < 8; ++r) {
        const int f = f0 + wm * 32 + i * 16 + mrow + r;
        if (f < NFRAMES)
          invf[(size_t)f * NFFT + nn] = acc[i][jj][r];
      }
    }
}

// ---------------- overlap-add + envelope + 1/N + unpad ----------------
__global__ void k_ola(const float* __restrict__ invf, float* __restrict__ out) {
  const int p = blockIdx.x * 256 + threadIdx.x;
  if (p >= LSIG) return;
  const int i = p + PAD;
  const int fmax = min(i >> 9, NFRAMES - 1);
  const int fmin = (i >= NFFT) ? ((i - NFFT + 1 + HOP - 1) >> 9) : 0;
  float s = 0.f;
  for (int f = fmin; f <= fmax; ++f)
    s += invf[(size_t)f * NFFT + (i - (f << 9))];
  out[p] = s / ((float)(fmax - fmin + 1) * (float)NFFT);
}

// ---------------- MSE (two-stage, deterministic) ----------------
__global__ void k_mse_partial(const float* __restrict__ xi, const float* __restrict__ yi,
                              float* __restrict__ partial) {
  __shared__ float sred[256];
  const int tid = threadIdx.x;
  float acc = 0.f;
  for (int p = blockIdx.x * 256 + tid; p < LSIG; p += 256 * 1024) {
    const float d = xi[p] - yi[p];
    acc += d * d;
  }
  sred[tid] = acc;
  __syncthreads();
  for (int s = 128; s > 0; s >>= 1) { if (tid < s) sred[tid] += sred[tid + s]; __syncthreads(); }
  if (tid == 0) partial[blockIdx.x] = sred[0];
}

__global__ void k_mse_final(const float* __restrict__ partial, float* __restrict__ out) {
  __shared__ float sred[256];
  const int tid = threadIdx.x;
  float acc = 0.f;
  for (int p = tid; p < 1024; p += 256) acc += partial[p];
  sred[tid] = acc;
  __syncthreads();
  for (int s = 128; s > 0; s >>= 1) { if (tid < s) sred[tid] += sred[tid + s]; __syncthreads(); }
  if (tid == 0) out[0] = sred[0] * (1.f / (float)LSIG);
}

// ---------------- launch ----------------
extern "C" void kernel_launch(void* const* d_in, const int* in_sizes, int n_in,
                              void* d_out, int out_size, void* d_ws, size_t ws_size,
                              hipStream_t stream) {
  (void)in_sizes; (void)n_in; (void)out_size; (void)ws_size;
  const float* x = (const float*)d_in[0];
  const float* y = (const float*)d_in[1];
  float* out = (float*)d_out;

  char* p = (char*)d_ws;
  auto carve = [&](size_t bytes) -> void* {
    void* r = (void*)p;
    p += (bytes + 255) & ~(size_t)255;
    return r;
  };
  _Float16* CfS    = (_Float16*)carve((size_t)NFFT * NPAD * 2);
  _Float16* CiS    = (_Float16*)carve((size_t)NPAD * NFFT * 2);
  int*      filt   = (int*)carve(NBINS * 4);
  int*      fstart = (int*)carve((NFILTS + 2) * 4);
  float*    scale  = (float*)carve(NFILTS * 4);
  float*    bq     = (float*)carve(NFILTS * 4);
  float*    sprA   = (float*)carve(NFILTS * NFILTS * 4);
  float*    spec   = (float*)carve((size_t)NFRAMES * NPAD * 4);
  _Float16* masked = (_Float16*)carve((size_t)NFRAMES * NPAD * 2);
  float*    invf   = (float*)carve((size_t)NFRAMES * NFFT * 4);
  float*    xi     = (float*)carve((size_t)LSIG * 4);
  float*    yi     = (float*)carve((size_t)LSIG * 4);
  float*    part   = (float*)carve(1024 * 4);

  k_build_mats<<<(NFFT * NPAD + 255) / 256, 256, 0, stream>>>(CfS, CiS);
  k_build_bark<<<1, 128, 0, stream>>>(filt, fstart, scale, bq, sprA);

  const float* sigs[2] = {x, y};
  float* outs[2] = {xi, yi};
  for (int s = 0; s < 2; ++s) {
    k_gemm_fwd<<<dim3(NPAD / 64, (NFRAMES + 127) / 128), 256, 0, stream>>>(sigs[s], CfS, spec);
    k_mask<<<NFRAMES, 128, 0, stream>>>(spec, masked, filt, fstart, scale, bq, sprA);
    k_gemm_inv<<<dim3(NFFT / 64, (NFRAMES + 127) / 128), 256, 0, stream>>>(masked, CiS, invf);
    k_ola<<<(LSIG + 255) / 256, 256, 0, stream>>>(invf, outs[s]);
  }
  k_mse_partial<<<1024, 256, 0, stream>>>(xi, yi, part);
  k_mse_final<<<1, 256, 0, stream>>>(part, out);
}
